// QuantLinearWithStats_81260781240394
// MI455X (gfx1250) — matrix operand order repaired and checked by reading, not compiled, not run on hardware
//
#include <hip/hip_runtime.h>

// ---------------------------------------------------------------------------
// Fused ternary-quantize + GEMM for MI455X (gfx1250, wave32, WMMA).
//   out[m,n] = sum_k  tern(x[m,k];0.33,0.5) * ( tern(w[n,k];0.2,0.6) + rv[n,k] )
// M=16384, N=2048, K=2048. Memory-bound (~288 MiB min traffic @ 23.3 TB/s),
// so quantization is fused into the GEMM (single pass over x), weights live
// in L2 (32 MiB << 192 MB), and math runs on v_wmma_f32_16x16x32_f16 with
// f32 accumulation (x_q exact in f16, w_rram rel. error ~5e-4).
// ---------------------------------------------------------------------------

typedef _Float16 v16h __attribute__((ext_vector_type(16)));
typedef _Float16 v8h  __attribute__((ext_vector_type(8)));
typedef _Float16 v4h  __attribute__((ext_vector_type(4)));
typedef float    v8f  __attribute__((ext_vector_type(8)));
typedef float    v4f  __attribute__((ext_vector_type(4)));

#define BM 128
#define BN 128
#define BK 32
#define LDT 40   // LDS row stride in halves: 32 data + 8 pad (80B => conflict-free b128)

__global__ __launch_bounds__(256)
void ternary_wmma_gemm(const float* __restrict__ x,
                       const float* __restrict__ w,
                       const float* __restrict__ rv,
                       float* __restrict__ out)
{
    constexpr int M = 4 * 4096;
    constexpr int N = 2048;
    constexpr int K = 2048;

    __shared__ _Float16 As[BM * LDT];   // x_q tile,  [128 rows(M)][32 k] f16
    __shared__ _Float16 Ws[BN * LDT];   // w_rram tile,[128 rows(N)][32 k] f16

    const int tid  = threadIdx.x;
    const int lane = tid & 31;
    const int wave = tid >> 5;
    const int wm   = wave >> 2;     // 0..1  -> 64 M-rows per wave
    const int wn   = wave & 3;      // 0..3  -> 32 N-cols per wave
    const int ln   = lane & 15;     // row/col within 16x16 fragment
    const int h    = lane >> 4;     // half-select per CDNA5 WMMA layout

    const int mBase = blockIdx.y * BM;
    const int nBase = blockIdx.x * BN;

    v8f acc[4][2] = {};             // eight 16x16 f32 accumulators per wave

    for (int kt = 0; kt < K; kt += BK) {
        __syncthreads();            // LDS reuse guard (readers from prev iter)

        // ---- stage A tile: 128x32 fp32 -> ternary f16 in LDS --------------
        // 1024 float4 chunks, 4 per thread; coalesced b128 global loads.
        #pragma unroll
        for (int i = 0; i < 4; ++i) {
            const int f = tid + 256 * i;
            const int r = f >> 3;           // 0..127  (M within tile)
            const int c = (f & 7) << 2;     // 0,4,...,28 (K within tile)
            const v4f xv = *(const v4f*)(x + (size_t)(mBase + r) * K + kt + c);
            v4h q;
            #pragma unroll
            for (int e = 0; e < 4; ++e) {
                const float v = xv[e];
                q[e] = (_Float16)((v > 0.33f) ? 0.5f : ((v < -0.33f) ? -0.5f : 0.0f));
            }
            *(v4h*)&As[r * LDT + c] = q;    // ds_store_b64
        }

        // ---- stage W tile: tern(weight)+rram -> f16 in LDS ----------------
        #pragma unroll
        for (int i = 0; i < 4; ++i) {
            const int f = tid + 256 * i;
            const int r = f >> 3;           // 0..127  (N within tile)
            const int c = (f & 7) << 2;
            const size_t off = (size_t)(nBase + r) * K + kt + c;
            const v4f wv = *(const v4f*)(w  + off);
            const v4f vv = *(const v4f*)(rv + off);
            v4h q;
            #pragma unroll
            for (int e = 0; e < 4; ++e) {
                const float t = (wv[e] > 0.2f) ? 0.6f
                              : ((wv[e] < -0.2f) ? -0.6f : 0.0f);
                q[e] = (_Float16)(t + vv[e]);
            }
            *(v4h*)&Ws[r * LDT + c] = q;
        }

        __syncthreads();

        // ---- load fragments per documented CDNA5 WMMA VGPR layouts --------
        // A 16x32 f16: lane ln holds row M=ln; h=0 -> K 0..7 & 16..23,
        //              h=1 -> K 8..15 & 24..31  => two b128 per fragment.
        v16h afrag[4];
        #pragma unroll
        for (int mf = 0; mf < 4; ++mf) {
            const int row = wm * 64 + mf * 16 + ln;
            const v8h lo = *(const v8h*)&As[row * LDT + 8 * h];        // K 8h..8h+7
            const v8h hi = *(const v8h*)&As[row * LDT + 8 * h + 16];   // K 8h+16..8h+23
            v16h a;
            #pragma unroll
            for (int e = 0; e < 8; ++e) { a[e] = lo[e]; a[8 + e] = hi[e]; }
            afrag[mf] = a;
        }
        // B 32x16 f16: lane ln = column N; h=0 -> K 0..7 & 16..23, h=1 -> K 8..15 & 24..31
        // (weights are [O,K] row-major, so this is contiguous in LDS too).
        v16h bfrag[2];
        #pragma unroll
        for (int nf = 0; nf < 2; ++nf) {
            const int col = wn * 32 + nf * 16 + ln;
            const v8h lo = *(const v8h*)&Ws[col * LDT + 8 * h];
            const v8h hi = *(const v8h*)&Ws[col * LDT + 8 * h + 16];
            v16h b;
            #pragma unroll
            for (int e = 0; e < 8; ++e) { b[e] = lo[e]; b[8 + e] = hi[e]; }
            bfrag[nf] = b;
        }

        // ---- 8 WMMAs per wave per K-step ---------------------------------
        #pragma unroll
        for (int mf = 0; mf < 4; ++mf)
            #pragma unroll
            for (int nf = 0; nf < 2; ++nf)
                acc[mf][nf] = __builtin_amdgcn_wmma_f32_16x16x32_f16(
                    /*neg_a=*/false, afrag[mf],
                    /*neg_b=*/false, bfrag[nf],
                    /*c_mod=*/(short)0, acc[mf][nf],
                    /*reuse_a=*/false, /*reuse_b=*/false);
    }

    // ---- store D: lane L -> (m = r + 8*(L>>4), n = L&15) per f32 C/D layout
    #pragma unroll
    for (int mf = 0; mf < 4; ++mf) {
        const int rowBase = mBase + wm * 64 + mf * 16 + 8 * h;
        #pragma unroll
        for (int nf = 0; nf < 2; ++nf) {
            const int col = nBase + wn * 32 + nf * 16 + ln;
            #pragma unroll
            for (int r = 0; r < 8; ++r)
                out[(size_t)(rowBase + r) * N + col] = acc[mf][nf][r];
        }
    }
}

extern "C" void kernel_launch(void* const* d_in, const int* in_sizes, int n_in,
                              void* d_out, int out_size, void* d_ws, size_t ws_size,
                              hipStream_t stream) {
    const float* x  = (const float*)d_in[0];   // [4,4096,2048]
    const float* w  = (const float*)d_in[1];   // [2048,2048]
    const float* rv = (const float*)d_in[2];   // [2048,2048]
    float* out = (float*)d_out;                // [4,4096,2048]

    constexpr int M = 4 * 4096;
    constexpr int N = 2048;
    dim3 grid(N / BN, M / BM);                 // 16 x 128 = 2048 blocks
    ternary_wmma_gemm<<<grid, 256, 0, stream>>>(x, w, rv, out);
}
